// CorrNet_13846974562742
// MI455X (gfx1250) — matrix operand order repaired and checked by reading, not compiled, not run on hardware
//
#include <hip/hip_runtime.h>
#include <hip/hip_bf16.h>

typedef __attribute__((ext_vector_type(8)))  __bf16 v8bf;
typedef __attribute__((ext_vector_type(16))) __bf16 v16bf;
typedef __attribute__((ext_vector_type(8)))  float  v8f;

// Global-address-space pointer types: force global_load_b128 (not flat_load)
// for the GEMM fragment loads regardless of infer-address-space limitations.
typedef const __attribute__((address_space(1))) __bf16 GBF;
typedef const __attribute__((address_space(1))) v8bf  GV8;

#define HW   4096   // 64*64 spatial positions
#define C9   2304   // 256 channels * 9-tap unfold
#define NB   4      // batch
#define NPAD 320    // padded N for warp GEMMs (256 feat + 3 img + zero pad)

// ---------------------------------------------------------------------------
// Build X[b][row][pos] bf16: rows 0..255 = feat channels, 256..258 = bilinear
// 4x-downsampled image channels, 259..319 = zeros (pad for N-tiling).
// ---------------------------------------------------------------------------
__global__ __launch_bounds__(256)
void k_prep_x(const float* __restrict__ feat, const float* __restrict__ img,
              __bf16* __restrict__ X)
{
    const int pos = blockIdx.x * 256 + threadIdx.x;   // 0..4095
    const int row = blockIdx.y;                       // 0..319
    const int b   = blockIdx.z;
    float v = 0.f;
    if (row < 256) {
        v = feat[((size_t)b * 256 + row) * HW + pos];
    } else if (row < 259) {
        const int ch = row - 256;
        const int y = pos >> 6, x = pos & 63;
        const float* p = img + ((size_t)b * 3 + ch) * 65536;
        const int yb = 4 * y, xb = 4 * x;
        v = 0.25f * (p[(yb + 1) * 256 + xb + 1] + p[(yb + 1) * 256 + xb + 2] +
                     p[(yb + 2) * 256 + xb + 1] + p[(yb + 2) * 256 + xb + 2]);
    }
    X[((size_t)b * NPAD + row) * HW + pos] = (__bf16)v;
}

// ---------------------------------------------------------------------------
// unfold3 + center + L2-normalize. One block per (b,pos); thread c owns the 9
// neighbor taps of channel c. Output position-major bf16: F[b][pos][k*256+c].
// (Element order within the 2304-vector is a fixed permutation of the
// reference's c*9+k order; mean/norm/dot-products are permutation invariant.)
// ---------------------------------------------------------------------------
__global__ __launch_bounds__(256)
void k_normalize(const float* __restrict__ feat, __bf16* __restrict__ out)
{
    __shared__ float red[256];
    const int pos = blockIdx.x, b = blockIdx.y;
    const int y = pos >> 6, x = pos & 63;
    const int c = threadIdx.x;
    const float* f = feat + (size_t)b * 256 * HW;

    float v[9];
    float s = 0.f;
#pragma unroll
    for (int k = 0; k < 9; ++k) {
        const int yy = y + k / 3 - 1, xx = x + k % 3 - 1;
        float val = 0.f;
        if ((unsigned)yy < 64u && (unsigned)xx < 64u)
            val = f[(size_t)c * HW + yy * 64 + xx];
        v[k] = val; s += val;
    }
    red[c] = s; __syncthreads();
    for (int st = 128; st > 0; st >>= 1) {
        if (c < st) red[c] += red[c + st];
        __syncthreads();
    }
    const float mean = red[0] * (1.f / (float)C9);
    __syncthreads();

    float sq = 0.f;
#pragma unroll
    for (int k = 0; k < 9; ++k) { v[k] -= mean; sq += v[k] * v[k]; }
    red[c] = sq; __syncthreads();
    for (int st = 128; st > 0; st >>= 1) {
        if (c < st) red[c] += red[c + st];
        __syncthreads();
    }
    const float inv = 1.f / (sqrtf(red[0]) + 1e-8f);

    __bf16* o = out + ((size_t)b * HW + pos) * C9;
#pragma unroll
    for (int k = 0; k < 9; ++k) o[k * 256 + c] = (__bf16)(v[k] * inv);
}

// ---------------------------------------------------------------------------
// Generic bf16 WMMA GEMM.  D[m][n] = sum_k A[m][k] * B[n][k]  (B row = column n,
// k contiguous -> clean 32B/lane B fragments; A row-major, k contiguous).
// Block = 256 threads = 8 waves; tile = 128(M) x 64(N); K unrolled by 64:
// 8 v_wmma_f32_16x16x32_bf16 per wave per iteration, single basic block.
// All fragment pointers are address_space(1) so loads/prefetches lower to
// global_load_b128 / global_prefetch_b8 (LOADcnt only, no flat/DScnt coupling).
// Optional outputs: out0 fp32 [m][n], out1 fp32 [n][m], out2 bf16 [n][m].
// ---------------------------------------------------------------------------
__global__ __launch_bounds__(256)
void k_gemm(const __bf16* __restrict__ A, size_t lda, size_t aBatch,
            const __bf16* __restrict__ B, size_t ldb, size_t bBatch,
            int K,
            float*  __restrict__ out0, size_t ld0, size_t o0Batch,
            float*  __restrict__ out1, size_t ld1, size_t o1Batch,
            __bf16* __restrict__ out2, size_t ld2, size_t o2Batch)
{
    const int tid  = threadIdx.x;
    const int wave = tid >> 5;
    const int lane = tid & 31;
    const int half = lane >> 4;   // 0: lanes 0-15, 1: lanes 16-31
    const int l16  = lane & 15;
    const int b    = blockIdx.z;
    const int i0   = blockIdx.x * 128 + wave * 16;   // wave's M base
    const int j0   = blockIdx.y * 64;                // block's N base

    // Per-lane base pointers in explicit global address space.
    GBF* ap  = (GBF*)(A + aBatch * b + (size_t)(i0 + l16) * lda) + half * 16;
    GBF* bp0 = (GBF*)(B + bBatch * b + (size_t)(j0 +  0 + l16) * ldb) + half * 16;
    GBF* bp1 = (GBF*)(B + bBatch * b + (size_t)(j0 + 16 + l16) * ldb) + half * 16;
    GBF* bp2 = (GBF*)(B + bBatch * b + (size_t)(j0 + 32 + l16) * ldb) + half * 16;
    GBF* bp3 = (GBF*)(B + bBatch * b + (size_t)(j0 + 48 + l16) * ldb) + half * 16;

    v8f acc[4] = {v8f{}, v8f{}, v8f{}, v8f{}};

    for (int k0 = 0; k0 < K; k0 += 64) {
#pragma unroll
        for (int u = 0; u < 2; ++u) {
            const int kk = u * 32;
            // A fragment: 16x32 bf16. lanes 0-15: K 0-15; lanes 16-31: 16-31
            v8bf alo = *(GV8*)(ap + kk);
            v8bf ahi = *(GV8*)(ap + kk + 8);
            v16bf af = __builtin_shufflevector(alo, ahi,
                        0,1,2,3,4,5,6,7,8,9,10,11,12,13,14,15);
            // Speculative prefetch 2 K-slabs ahead (dropped if OOB; no guard,
            // keeps the loop a single basic block).
            __builtin_prefetch(ap + kk + 128, 0, 1);

            GBF* bps[4] = {bp0, bp1, bp2, bp3};
#pragma unroll
            for (int t = 0; t < 4; ++t) {
                // B fragment: 32x16 bf16. lane = col n; lanes 0-15: K 0-15, 16-31: 16-31
                v8bf blo = *(GV8*)(bps[t] + kk);
                v8bf bhi = *(GV8*)(bps[t] + kk + 8);
                v16bf bfm = __builtin_shufflevector(blo, bhi,
                             0,1,2,3,4,5,6,7,8,9,10,11,12,13,14,15);
                __builtin_prefetch(bps[t] + kk + 128, 0, 1);

                acc[t] = __builtin_amdgcn_wmma_f32_16x16x32_bf16(
                            false, af, false, bfm, (short)0, acc[t], false, false);
            }
        }
        ap += 64; bp0 += 64; bp1 += 64; bp2 += 64; bp3 += 64;
    }

    // C/D layout: VGPR r -> M = r + 8*half ; N = lane&15 (per 16x16 tile)
#pragma unroll
    for (int t = 0; t < 4; ++t) {
        const int n = j0 + t * 16 + l16;
#pragma unroll
        for (int r = 0; r < 8; ++r) {
            const int m = i0 + r + 8 * half;
            const float v = acc[t][r];
            if (out0) out0[o0Batch * b + (size_t)m * ld0 + n] = v;
            if (out1) out1[o1Batch * b + (size_t)n * ld1 + m] = v;
            if (out2) out2[o2Batch * b + (size_t)n * ld2 + m] = (__bf16)v;
        }
    }
}

// ---------------------------------------------------------------------------
// Row softmax with temperature 0.01 (logits * 100), fp32 in -> bf16 out.
// One block per (b,row); thread owns 16 strided elements.
// ---------------------------------------------------------------------------
__global__ __launch_bounds__(256)
void k_softmax(const float* __restrict__ corr, __bf16* __restrict__ P)
{
    __shared__ float red[256];
    const int row = blockIdx.x, b = blockIdx.y;
    const int tid = threadIdx.x;
    const float* r = corr + ((size_t)b * HW + row) * HW;

    float v[16];
    float m = -1e30f;
#pragma unroll
    for (int q = 0; q < 16; ++q) { v[q] = r[q * 256 + tid]; m = fmaxf(m, v[q]); }
    red[tid] = m; __syncthreads();
    for (int st = 128; st > 0; st >>= 1) {
        if (tid < st) red[tid] = fmaxf(red[tid], red[tid + st]);
        __syncthreads();
    }
    m = red[0]; __syncthreads();

    float s = 0.f;
#pragma unroll
    for (int q = 0; q < 16; ++q) { v[q] = __expf((v[q] - m) * 100.f); s += v[q]; }
    red[tid] = s; __syncthreads();
    for (int st = 128; st > 0; st >>= 1) {
        if (tid < st) red[tid] += red[tid + st];
        __syncthreads();
    }
    const float inv = 1.f / red[0];

    __bf16* o = P + ((size_t)b * HW + row) * HW;
#pragma unroll
    for (int q = 0; q < 16; ++q) o[q * 256 + tid] = (__bf16)(v[q] * inv);
}

// ---------------------------------------------------------------------------
// Assemble [4,777,64,64]: W1=P@Xs, W3=Q@W1, W4=P@W2 all stored [b][320][pos].
// Channels: 0-255 W1 feats | 256-511 W3 | 512-767 W4 | 768-770 W1 imgs
//           | 771-773 W3 imgs | 774-776 W4 imgs
// ---------------------------------------------------------------------------
__global__ __launch_bounds__(256)
void k_gather(const float* __restrict__ W1, const float* __restrict__ W3,
              const float* __restrict__ W4, float* __restrict__ out)
{
    const size_t idx = (size_t)blockIdx.x * 256 + threadIdx.x;
    const size_t total = (size_t)NB * 777 * HW;
    if (idx >= total) return;
    const int pos = (int)(idx & (HW - 1));
    const size_t t = idx >> 12;
    const int ch = (int)(t % 777);
    const int b  = (int)(t / 777);

    const float* src; int r;
    if      (ch < 256) { src = W1; r = ch; }
    else if (ch < 512) { src = W3; r = ch - 256; }
    else if (ch < 768) { src = W4; r = ch - 512; }
    else if (ch < 771) { src = W1; r = 256 + (ch - 768); }
    else if (ch < 774) { src = W3; r = 256 + (ch - 771); }
    else               { src = W4; r = 256 + (ch - 774); }
    out[idx] = src[((size_t)b * NPAD + r) * HW + pos];
}

// ---------------------------------------------------------------------------
extern "C" void kernel_launch(void* const* d_in, const int* in_sizes, int n_in,
                              void* d_out, int out_size, void* d_ws, size_t ws_size,
                              hipStream_t stream)
{
    (void)in_sizes; (void)n_in; (void)out_size; (void)ws_size;
    const float* content = (const float*)d_in[0];
    const float* style   = (const float*)d_in[1];
    const float* c_feat  = (const float*)d_in[2];
    const float* s_feat  = (const float*)d_in[3];
    float* out = (float*)d_out;

    char* ws = (char*)d_ws;
    size_t off = 0;
    auto alloc = [&](size_t bytes) -> void* {
        void* p = ws + off;
        off += (bytes + 255) & ~(size_t)255;
        return p;
    };
    __bf16* Fc    = (__bf16*)alloc((size_t)NB * HW * C9 * 2);
    __bf16* Fs    = (__bf16*)alloc((size_t)NB * HW * C9 * 2);
    float*  corr  = (float*) alloc((size_t)NB * HW * HW * 4);
    float*  corrT = (float*) alloc((size_t)NB * HW * HW * 4);
    __bf16* Pbf   = (__bf16*)alloc((size_t)NB * HW * HW * 2);
    __bf16* Qbf   = (__bf16*)alloc((size_t)NB * HW * HW * 2);
    __bf16* Xs    = (__bf16*)alloc((size_t)NB * NPAD * HW * 2);
    __bf16* Xc    = (__bf16*)alloc((size_t)NB * NPAD * HW * 2);
    float*  W1f   = (float*) alloc((size_t)NB * NPAD * HW * 4);
    __bf16* W1b   = (__bf16*)alloc((size_t)NB * NPAD * HW * 2);
    __bf16* W2b   = (__bf16*)alloc((size_t)NB * NPAD * HW * 2);
    float*  W3f   = (float*) alloc((size_t)NB * NPAD * HW * 4);
    float*  W4f   = (float*) alloc((size_t)NB * NPAD * HW * 4);

    const dim3 blk(256);
    const size_t fB = (size_t)HW * C9;      // F batch stride (elems)
    const size_t cB = (size_t)HW * HW;      // corr batch stride
    const size_t xB = (size_t)NPAD * HW;    // X / W batch stride

    // Pack features + downsampled images to bf16 [b][320][4096]
    k_prep_x<<<dim3(16, NPAD, NB), blk, 0, stream>>>(s_feat, style, Xs);
    k_prep_x<<<dim3(16, NPAD, NB), blk, 0, stream>>>(c_feat, content, Xc);

    // Unfold + center + L2-normalize -> position-major bf16
    k_normalize<<<dim3(HW, NB), blk, 0, stream>>>(c_feat, Fc);
    k_normalize<<<dim3(HW, NB), blk, 0, stream>>>(s_feat, Fs);

    // corr[i][j] and corr^T (so both softmaxes are row-major)
    k_gemm<<<dim3(32, 64, NB), blk, 0, stream>>>(
        Fc, C9, fB, Fs, C9, fB, C9,
        corr, HW, cB, corrT, HW, cB, nullptr, 0, 0);

    k_softmax<<<dim3(HW, NB), blk, 0, stream>>>(corr,  Pbf);
    k_softmax<<<dim3(HW, NB), blk, 0, stream>>>(corrT, Qbf);

    // W1 = P @ Xs  (fp32 for output + bf16 for chaining)
    k_gemm<<<dim3(32, 5, NB), blk, 0, stream>>>(
        Pbf, HW, cB, Xs, HW, xB, HW,
        nullptr, 0, 0, W1f, HW, xB, W1b, HW, xB);
    // W2 = Q @ Xc  (bf16 only)
    k_gemm<<<dim3(32, 5, NB), blk, 0, stream>>>(
        Qbf, HW, cB, Xc, HW, xB, HW,
        nullptr, 0, 0, nullptr, 0, 0, W2b, HW, xB);
    // W3 = Q @ W1
    k_gemm<<<dim3(32, 5, NB), blk, 0, stream>>>(
        Qbf, HW, cB, W1b, HW, xB, HW,
        nullptr, 0, 0, W3f, HW, xB, nullptr, 0, 0);
    // W4 = P @ W2
    k_gemm<<<dim3(32, 5, NB), blk, 0, stream>>>(
        Pbf, HW, cB, W2b, HW, xB, HW,
        nullptr, 0, 0, W4f, HW, xB, nullptr, 0, 0);

    // Assemble [4,777,64,64]
    const size_t total = (size_t)NB * 777 * HW;
    k_gather<<<dim3((unsigned)((total + 255) / 256)), blk, 0, stream>>>(
        W1f, W3f, W4f, out);
}
